// FullMHLA_7868380086610
// MI455X (gfx1250) — matrix operand order repaired and checked by reading, not compiled, not run on hardware
//
#include <hip/hip_runtime.h>
#include <hip/hip_bf16.h>

// ---------------------------------------------------------------------------
// MLA forward for MI455X (gfx1250, wave32, WMMA).
// - Every heavy op is a batched f16->f32 WMMA GEMM (v_wmma_f32_16x16x32_f16).
// - All GEMMs consume B in (N x K) layout so LDS staging is pure b128 copies.
// - Staging uses CDNA5 GLOBAL_LOAD_ASYNC_TO_LDS_B128 (inline asm, ASYNCcnt)
//   with a 3-stage LDS pipeline; steady-state loop is branch-free: the two
//   tail iterations are peeled so s_wait_asynccnt thresholds are static.
// - Attention matrix P (64MB f16) is materialized: it fits the 192MB L2, so
//   S -> softmax -> P round trips are negligible vs ~110 GFLOP of matmul.
// ---------------------------------------------------------------------------

typedef __attribute__((ext_vector_type(16))) _Float16 v16h;
typedef __attribute__((ext_vector_type(8)))  float    v8f;

union F16V { v16h v; uint4 q[2]; };

#define LDSTRIDE 40                         // 32 K + 8 pad halves = 80B rows
constexpr int A_ELEMS = 128 * LDSTRIDE;     // A stage tile (128 rows x 32 K)
constexpr int B_ELEMS = 64 * LDSTRIDE;      // B stage tile (64 rows  x 32 K)
constexpr int STAGE_ELEMS = A_ELEMS + B_ELEMS;

__device__ __forceinline__ void async_b128(unsigned ldsaddr, const void* gsrc) {
    // GLOBAL_LOAD_ASYNC_TO_LDS_B128: copies 16B from global memory to the LDS address in vdst, tracked by ASYNCcnt
    asm volatile("global_load_async_to_lds_b128 %0, %1, off"
                 :: "v"(ldsaddr), "v"((unsigned long long)(uintptr_t)gsrc)
                 : "memory");
}

// Batched GEMM: C[z] (MxN) = A[z] (MxK, row-major) * Bt[z]^T, where Bt is
// stored (N x K) row-major (K contiguous). Block tile 128x64, 8 waves, each
// wave 32x32 via 2x2 WMMA fragments, K stepped by 32.
template<bool F32OUT>
__global__ __launch_bounds__(256) void gemm_wmma_kernel(
    const _Float16* __restrict__ A, long lda, long sA,
    const _Float16* __restrict__ Bt, long ldb, long sB,
    void* __restrict__ Cv, long ldc, long sC, int K)
{
    __shared__ __align__(16) _Float16 lds[3 * STAGE_ELEMS];

    const int z = blockIdx.z;
    A  += (long)z * sA;
    Bt += (long)z * sB;

    const long m0 = (long)blockIdx.y * 128;
    const long n0 = (long)blockIdx.x * 64;
    const int tid  = threadIdx.x;
    const int lane = tid & 31;
    const int wv   = tid >> 5;
    const int m_w  = (wv & 3) * 32;     // 4 row groups of 32
    const int n_w  = (wv >> 2) * 32;    // 2 col groups of 32
    const int rl   = lane & 15;
    const int kb   = (lane >> 4) * 8;   // A frag: K base within 16-half
    const int kh   = (lane >> 4) * 16;  // unused: B frag reads with kb, same as A

    // staging split across 256 threads (3 async b128 per thread per stage)
    const int a_row = tid >> 1, a_cb = (tid & 1) * 16;  // A: 128 rows x 32 K
    const int b_n   = tid >> 2, b_kb = (tid & 3) * 8;   // B: 64  rows x 32 K

    const unsigned lds_base = (unsigned)(unsigned long long)(uintptr_t)(&lds[0]);
    const _Float16* aRow = A  + (m0 + a_row) * lda + a_cb;
    const _Float16* bRow = Bt + (n0 + b_n)  * ldb + b_kb;

    auto stage = [&](int s, int buf) {
        const long k0 = (long)s << 5;
        unsigned ad = lds_base + (unsigned)((buf * STAGE_ELEMS + a_row * LDSTRIDE + a_cb) * 2);
        async_b128(ad,       aRow + k0);
        async_b128(ad + 16u, aRow + k0 + 8);
        unsigned bd = lds_base + (unsigned)((buf * STAGE_ELEMS + A_ELEMS + b_n * LDSTRIDE + b_kb) * 2);
        async_b128(bd, bRow + k0);
    };

    v8f acc00 = {}, acc01 = {}, acc10 = {}, acc11 = {};

    auto compute = [&](int buf) {
        const _Float16* sAh = lds + buf * STAGE_ELEMS;
        const _Float16* sBh = sAh + A_ELEMS;
        F16V a0, a1, b0, b1;
        const _Float16* ra0 = sAh + (m_w + rl)      * LDSTRIDE;
        const _Float16* ra1 = sAh + (m_w + 16 + rl) * LDSTRIDE;
        const _Float16* rb0 = sBh + (n_w + rl)      * LDSTRIDE;
        const _Float16* rb1 = sBh + (n_w + 16 + rl) * LDSTRIDE;
        a0.q[0] = *(const uint4*)(ra0 + kb);  a0.q[1] = *(const uint4*)(ra0 + 16 + kb);
        a1.q[0] = *(const uint4*)(ra1 + kb);  a1.q[1] = *(const uint4*)(ra1 + 16 + kb);
        b0.q[0] = *(const uint4*)(rb0 + kb);  b0.q[1] = *(const uint4*)(rb0 + 16 + kb);
        b1.q[0] = *(const uint4*)(rb1 + kb);  b1.q[1] = *(const uint4*)(rb1 + 16 + kb);
        acc00 = __builtin_amdgcn_wmma_f32_16x16x32_f16(false, a0.v, false, b0.v, (short)0, acc00, false, false);
        acc01 = __builtin_amdgcn_wmma_f32_16x16x32_f16(false, a0.v, false, b1.v, (short)0, acc01, false, false);
        acc10 = __builtin_amdgcn_wmma_f32_16x16x32_f16(false, a1.v, false, b0.v, (short)0, acc10, false, false);
        acc11 = __builtin_amdgcn_wmma_f32_16x16x32_f16(false, a1.v, false, b1.v, (short)0, acc11, false, false);
    };

    const int nsteps = K >> 5;          // >= 4 and even for every GEMM here
    stage(0, 0);
    if (nsteps > 1) stage(1, 1);

    int bi = 0;  // LDS buffer holding stage i
    // Steady state: stage i+1 in flight (3 asyncs); in-order retirement means
    // "<=3 outstanding" implies stage i has fully landed.
    #pragma unroll 2
    for (int i = 0; i < nsteps - 2; ++i) {
        asm volatile("s_wait_asynccnt 0x3" ::: "memory");
        __syncthreads();
        compute(bi);
        stage(i + 2, (bi + 2 >= 3) ? (bi - 1) : (bi + 2));  // overwrites stage i-1's buffer
        bi = (bi + 1 == 3) ? 0 : bi + 1;
    }
    if (nsteps >= 2) {                   // iteration nsteps-2: nothing left to stage
        asm volatile("s_wait_asynccnt 0x3" ::: "memory");
        __syncthreads();
        compute(bi);
        bi = (bi + 1 == 3) ? 0 : bi + 1;
    }
    // iteration nsteps-1: drain fully
    asm volatile("s_wait_asynccnt 0x0" ::: "memory");
    __syncthreads();
    compute(bi);

    // store per ISA C/D layout: VGPR i -> M=(lane>=16?8:0)+i, lane&15 -> N
    const long col0 = n0 + n_w + rl;
    const long row0 = m0 + m_w + (lane >> 4) * 8;
    if (F32OUT) {
        float* Cf = (float*)Cv + (long)z * sC;
        #pragma unroll
        for (int i = 0; i < 8; ++i) {
            long r0 = row0 + i, r1 = row0 + 16 + i;
            Cf[r0 * ldc + col0]      = acc00[i];
            Cf[r0 * ldc + col0 + 16] = acc01[i];
            Cf[r1 * ldc + col0]      = acc10[i];
            Cf[r1 * ldc + col0 + 16] = acc11[i];
        }
    } else {
        _Float16* Ch = (_Float16*)Cv + (long)z * sC;
        #pragma unroll
        for (int i = 0; i < 8; ++i) {
            long r0 = row0 + i, r1 = row0 + 16 + i;
            Ch[r0 * ldc + col0]      = (_Float16)acc00[i];
            Ch[r0 * ldc + col0 + 16] = (_Float16)acc01[i];
            Ch[r1 * ldc + col0]      = (_Float16)acc10[i];
            Ch[r1 * ldc + col0 + 16] = (_Float16)acc11[i];
        }
    }
}

// ---------------------------------------------------------------------------
// Elementwise / layout kernels
// ---------------------------------------------------------------------------
__global__ void cast_f32_to_f16(const float* __restrict__ in, _Float16* __restrict__ out, int n) {
    int i = blockIdx.x * blockDim.x + threadIdx.x;
    if (i < n) out[i] = (_Float16)in[i];
}

// in[z]: (R x Cc) row-major -> out[z]: (Cc x R) row-major (f16), z = blockIdx.y
template<typename TIN>
__global__ void transpose_to_f16(const TIN* __restrict__ in, _Float16* __restrict__ out,
                                 int R, int Cc, long inBS, long outBS) {
    int i = blockIdx.x * blockDim.x + threadIdx.x;
    if (i >= R * Cc) return;
    const TIN* ib = in + (long)blockIdx.y * inBS;
    _Float16* ob = out + (long)blockIdx.y * outBS;
    int r = i / Cc, c = i % Cc;
    ob[(long)c * R + r] = (_Float16)ib[i];
}

// RoPE q_r into qpack[..., 512:576]; one thread per (b,t,h,pair)
__global__ void rope_q_kernel(const _Float16* __restrict__ c_qr, const float* __restrict__ fc,
                              const float* __restrict__ fs, _Float16* __restrict__ qpack, int n) {
    int i = blockIdx.x * blockDim.x + threadIdx.x;
    if (i >= n) return;
    int j = i & 31, h = (i >> 5) & 15, t = (i >> 9) & 1023, b = i >> 19;
    const _Float16* src = c_qr + ((long)(b * 1024 + t)) * 1024 + h * 64 + 2 * j;
    float re = (float)src[0], im = (float)src[1];
    float c = fc[t * 32 + j], s = fs[t * 32 + j];
    long o = ((long)((b * 16 + h) * 1024 + t)) * 576 + 512 + 2 * j;
    qpack[o]     = (_Float16)(re * c - im * s);
    qpack[o + 1] = (_Float16)(re * s + im * c);
}

__global__ void pack_kv_kernel(const _Float16* __restrict__ c_kv, _Float16* __restrict__ kpack, int n) {
    int i = blockIdx.x * blockDim.x + threadIdx.x;
    if (i >= n) return;
    int row = i >> 9, col = i & 511;
    kpack[(long)row * 576 + col] = c_kv[i];
}

__global__ void rope_k_kernel(const _Float16* __restrict__ c_kr, const float* __restrict__ fc,
                              const float* __restrict__ fs, _Float16* __restrict__ kpack, int n) {
    int i = blockIdx.x * blockDim.x + threadIdx.x;
    if (i >= n) return;
    int row = i >> 5, j = i & 31, t = row & 1023;
    const _Float16* src = c_kr + (long)row * 64 + 2 * j;
    float re = (float)src[0], im = (float)src[1];
    float c = fc[t * 32 + j], s = fs[t * 32 + j];
    long o = (long)row * 576 + 512 + 2 * j;
    kpack[o]     = (_Float16)(re * c - im * s);
    kpack[o + 1] = (_Float16)(re * s + im * c);
}

// Causal softmax over one 1024-wide row per block; scale 1/sqrt(192).
// Masked columns written as 0 so the following GEMM can use full K=1024.
__global__ __launch_bounds__(256) void softmax_causal(_Float16* __restrict__ P) {
    __shared__ float red[256];
    const float scale = 0.07216878364870323f;   // 1/sqrt(128+64)
    int rowid = blockIdx.x;                      // (b*16+h)*1024 + t
    int t = rowid & 1023;
    _Float16* p = P + (long)rowid * 1024;
    int tid = threadIdx.x;

    float m = -3.0e38f;
    for (int c = tid; c <= t; c += 256) m = fmaxf(m, (float)p[c] * scale);
    red[tid] = m;
    for (int s = 128; s > 0; s >>= 1) { __syncthreads(); if (tid < s) red[tid] = fmaxf(red[tid], red[tid + s]); }
    __syncthreads();
    m = red[0];
    __syncthreads();

    float sum = 0.f;
    for (int c = tid; c <= t; c += 256) sum += __expf((float)p[c] * scale - m);
    red[tid] = sum;
    for (int s = 128; s > 0; s >>= 1) { __syncthreads(); if (tid < s) red[tid] += red[tid + s]; }
    __syncthreads();
    float inv = 1.0f / red[0];

    for (int c = tid; c < 1024; c += 256) {
        float v = (c <= t) ? __expf((float)p[c] * scale - m) * inv : 0.0f;
        p[c] = (_Float16)v;
    }
}

// ---------------------------------------------------------------------------
// Host side
// ---------------------------------------------------------------------------
static inline void launch_gemm(bool f32out,
                               const _Float16* A, long lda, long sA,
                               const _Float16* Bt, long ldb, long sB,
                               void* Cp, long ldc, long sC,
                               int M, int N, int K, int batch, hipStream_t st)
{
    dim3 g(N / 64, M / 128, batch), blk(256, 1, 1);
    if (f32out)
        gemm_wmma_kernel<true ><<<g, blk, 0, st>>>(A, lda, sA, Bt, ldb, sB, Cp, ldc, sC, K);
    else
        gemm_wmma_kernel<false><<<g, blk, 0, st>>>(A, lda, sA, Bt, ldb, sB, Cp, ldc, sC, K);
}

extern "C" void kernel_launch(void* const* d_in, const int* in_sizes, int n_in,
                              void* d_out, int out_size, void* d_ws, size_t ws_size,
                              hipStream_t stream) {
    (void)in_sizes; (void)n_in; (void)out_size; (void)ws_size;

    constexpr int Bb = 2, T = 1024, Cc = 2048, NH = 16, NLQ = 512, NLKV = 512, DHR = 64, HS = 128;
    constexpr int BT = Bb * T;          // 2048
    constexpr int DPK = NLKV + DHR;     // 576 packed QK width

    const float* x     = (const float*)d_in[0];
    const float* fcos  = (const float*)d_in[1];
    const float* fsin  = (const float*)d_in[2];
    const float* W_dq  = (const float*)d_in[3];
    const float* W_uq  = (const float*)d_in[4];
    const float* W_dkv = (const float*)d_in[5];
    const float* W_uk  = (const float*)d_in[6];
    const float* W_uv  = (const float*)d_in[7];
    const float* W_qr  = (const float*)d_in[8];
    const float* W_kr  = (const float*)d_in[9];
    const float* W_o   = (const float*)d_in[10];
    float* out = (float*)d_out;

    // ---- workspace bump allocator (f16 buffers, 256B aligned)
    char* base = (char*)d_ws;
    size_t off = 0;
    auto alloc16 = [&](size_t elems) -> _Float16* {
        _Float16* p = (_Float16*)(base + off);
        off += ((elems * sizeof(_Float16) + 255) / 256) * 256;
        return p;
    };
    _Float16* x_h    = alloc16((size_t)BT * Cc);
    _Float16* Wdq_h  = alloc16((size_t)NLQ * Cc);       // (512 x 2048) = Bt for c_q
    _Float16* Wdkv_h = alloc16((size_t)NLKV * Cc);
    _Float16* Wkr_h  = alloc16((size_t)DHR * Cc);
    _Float16* Wqr_h  = alloc16((size_t)NH * DHR * NLQ);
    _Float16* Wuq_h  = alloc16((size_t)Cc * NLQ);       // flat (q,h,d)
    _Float16* Wo_h   = alloc16((size_t)Cc * Cc);
    _Float16* Wuv_t  = alloc16((size_t)NLKV * Cc);      // W_uv^T (512 x 2048)
    _Float16* WukT   = alloc16((size_t)NH * NLKV * HS); // per-head (kk x d)
    _Float16* keffT  = alloc16((size_t)NH * NLKV * NLQ);// per-head (kk x q)
    _Float16* veffT  = alloc16((size_t)Cc * NLKV);      // (c x k) = W_o @ W_uv
    _Float16* cq     = alloc16((size_t)BT * NLQ);
    _Float16* ckv    = alloc16((size_t)BT * NLKV);
    _Float16* ckvT   = alloc16((size_t)Bb * NLKV * T);  // per-b (ck x s)
    _Float16* ckr    = alloc16((size_t)BT * DHR);
    _Float16* cqr    = alloc16((size_t)BT * NH * DHR);
    _Float16* qpack  = alloc16((size_t)Bb * NH * T * DPK);
    _Float16* kpack  = alloc16((size_t)BT * DPK);
    _Float16* P      = alloc16((size_t)Bb * NH * T * T);
    _Float16* lat    = alloc16((size_t)Bb * NH * T * NLKV);

    const int TPB = 256;
    auto nb = [](int n) { return (n + 255) / 256; };

    // ---- f16 casts (weights used as Bt keep their natural row-major layout)
    cast_f32_to_f16<<<nb(BT * Cc),        TPB, 0, stream>>>(x,     x_h,    BT * Cc);
    cast_f32_to_f16<<<nb(NLQ * Cc),       TPB, 0, stream>>>(W_dq,  Wdq_h,  NLQ * Cc);
    cast_f32_to_f16<<<nb(NLKV * Cc),      TPB, 0, stream>>>(W_dkv, Wdkv_h, NLKV * Cc);
    cast_f32_to_f16<<<nb(DHR * Cc),       TPB, 0, stream>>>(W_kr,  Wkr_h,  DHR * Cc);
    cast_f32_to_f16<<<nb(NH * DHR * NLQ), TPB, 0, stream>>>(W_qr,  Wqr_h,  NH * DHR * NLQ);
    cast_f32_to_f16<<<nb(Cc * NLQ),       TPB, 0, stream>>>(W_uq,  Wuq_h,  Cc * NLQ);
    cast_f32_to_f16<<<nb(Cc * Cc),        TPB, 0, stream>>>(W_o,   Wo_h,   Cc * Cc);

    // ---- the two transposes that are genuinely needed
    transpose_to_f16<float><<<dim3(nb(Cc * NLKV), 1), TPB, 0, stream>>>(
        W_uv, Wuv_t, Cc, NLKV, 0, 0);
    transpose_to_f16<float><<<dim3(nb(HS * NLKV), NH), TPB, 0, stream>>>(
        W_uk, WukT, HS, NLKV, (long)HS * NLKV, (long)NLKV * HS);

    // ---- weight-only precompute
    // keffT[h] (kk x q) = WukT[h] (kk x d) @ Wuq[:,h,:]^T   (Bt rows q, contiguous d)
    launch_gemm(false, WukT, HS, (long)NLKV * HS,
                Wuq_h, Cc, HS,
                keffT, NLQ, (long)NLKV * NLQ, NLKV, NLQ, HS, NH, stream);
    // veffT (c x k) = W_o @ W_uv   (Bt = Wuv_t rows k, contiguous j)
    launch_gemm(false, Wo_h, Cc, 0, Wuv_t, Cc, 0,
                veffT, NLKV, 0, Cc, NLKV, Cc, 1, stream);

    // ---- projections: x @ W^T -> Bt is the raw weight
    launch_gemm(false, x_h, Cc, 0, Wdq_h,  Cc, 0, cq,  NLQ,      0, BT, NLQ,      Cc,  1, stream);
    launch_gemm(false, x_h, Cc, 0, Wdkv_h, Cc, 0, ckv, NLKV,     0, BT, NLKV,     Cc,  1, stream);
    launch_gemm(false, x_h, Cc, 0, Wkr_h,  Cc, 0, ckr, DHR,      0, BT, DHR,      Cc,  1, stream);
    launch_gemm(false, cq, NLQ, 0, Wqr_h, NLQ, 0, cqr, NH * DHR, 0, BT, NH * DHR, NLQ, 1, stream);

    // ---- q_abs straight into packed-Q (width 576), batched over heads per b
    for (int b = 0; b < Bb; ++b) {
        launch_gemm(false, cq + (long)b * T * NLQ, NLQ, 0,
                    keffT, NLQ, (long)NLKV * NLQ,
                    qpack + (long)b * NH * T * DPK, DPK, (long)T * DPK,
                    T, NLKV, NLQ, NH, stream);
    }
    // RoPE tails + packed K
    rope_q_kernel<<<nb(Bb * T * NH * 32), TPB, 0, stream>>>(cqr, fcos, fsin, qpack, Bb * T * NH * 32);
    pack_kv_kernel<<<nb(BT * NLKV),       TPB, 0, stream>>>(ckv, kpack, BT * NLKV);
    rope_k_kernel<<<nb(BT * 32),          TPB, 0, stream>>>(ckr, fcos, fsin, kpack, BT * 32);

    // ---- attention scores S = Qpack @ Kpack^T (K = 576), Bt = kpack rows s
    for (int b = 0; b < Bb; ++b) {
        launch_gemm(false, qpack + (long)b * NH * T * DPK, DPK, (long)T * DPK,
                    kpack + (long)b * T * DPK, DPK, 0,
                    P + (long)b * NH * T * T, T, (long)T * T,
                    T, T, DPK, NH, stream);
    }
    softmax_causal<<<Bb * NH * T, TPB, 0, stream>>>(P);

    // ---- lat = P @ c_kv : Bt = ckvT[b] (ck x s)
    transpose_to_f16<_Float16><<<dim3(nb(T * NLKV), Bb), TPB, 0, stream>>>(
        ckv, ckvT, T, NLKV, (long)T * NLKV, (long)NLKV * T);
    for (int b = 0; b < Bb; ++b) {
        launch_gemm(false, P + (long)b * NH * T * T, T, (long)T * T,
                    ckvT + (long)b * NLKV * T, T, 0,
                    lat + (long)b * NH * T * NLKV, NLKV, (long)T * NLKV,
                    T, NLKV, T, NH, stream);
    }
    // ---- y[b,t,h,:] = lat[b,h,t,:] @ v_eff[:,h,:] : Bt = veffT rows h*128+d
    for (int b = 0; b < Bb; ++b) {
        launch_gemm(true, lat + (long)b * NH * T * NLKV, NLKV, (long)T * NLKV,
                    veffT + (long)0, NLKV, (long)HS * NLKV,
                    out + (long)b * T * Cc, Cc, HS,
                    T, HS, NLKV, NH, stream);
    }
}
